// GraphAttentionLayer_33801392620000
// MI455X (gfx1250) — matrix operand order repaired and checked by reading, not compiled, not run on hardware
//
#include <hip/hip_runtime.h>
#include <hip/hip_bf16.h>
#include <math.h>

typedef __attribute__((ext_vector_type(2)))  float    v2f;
typedef __attribute__((ext_vector_type(4)))  float    v4f;
typedef __attribute__((ext_vector_type(8)))  float    v8f;
typedef __attribute__((ext_vector_type(8)))  _Float16 v8h;
typedef __attribute__((ext_vector_type(16))) _Float16 v16h;

#define BATCH 8
#define NN    2048
#define FF    256
#define GAT_ALPHA 0.2f

__device__ __forceinline__ float lrelu(float x) { return x > 0.f ? x : GAT_ALPHA * x; }
__device__ __forceinline__ float sigm(float x)  { return 1.f / (1.f + __expf(-x)); }

// ---------------------------------------------------------------------------
// Kernel 1: h = X @ W  (X: 16384x256, W: 256x256) in full f32 WMMA 16x16x4
// (input-sensitive GEMM: stays fp32). Writes h32 (row major, for scores) and
// h16T (f16, [b][f][j] transposed, B-operand of the big f16 GEMM).
// One wave per 16x16 output tile; 8 waves per block.
// ---------------------------------------------------------------------------
__global__ __launch_bounds__(256) void k_gemm_h(const float* __restrict__ X,
                                                const float* __restrict__ W,
                                                float* __restrict__ h32,
                                                _Float16* __restrict__ h16T) {
    const int lane = threadIdx.x & 31;
    const int wv   = threadIdx.x >> 5;
    const int tm   = blockIdx.x * 8 + wv;   // 0..1023 (row tiles of 16384 rows)
    const int tn   = blockIdx.y;            // 0..15   (col tiles of 256 cols)
    const int half = lane >> 4;
    const int l15  = lane & 15;
    const int row  = tm * 16 + l15;         // A-matrix M index
    const int col  = tn * 16 + l15;         // B/D N index

    const float* __restrict__ xr = X + (size_t)row * FF;

    v8f c = {};
    for (int k0 = 0; k0 < FF; k0 += 4) {
        const int ka = k0 + half * 2;
        v2f a, b;
        a.x = xr[ka];
        a.y = xr[ka + 1];
        b.x = W[(size_t)ka * FF + col];
        b.y = W[(size_t)(ka + 1) * FF + col];
        c = __builtin_amdgcn_wmma_f32_16x16x4_f32(false, a, false, b,
                                                  (short)0, c, false, false);
    }

    const int mbase = tm * 16 + half * 8;
    #pragma unroll
    for (int r = 0; r < 8; ++r) {
        const int m = mbase + r;            // global row: b*NN + i
        h32[(size_t)m * FF + col] = c[r];
        const int b = m >> 11;
        const int i = m & (NN - 1);
        h16T[((size_t)(b * FF + col)) * NN + i] = (_Float16)c[r];
    }
}

// ---------------------------------------------------------------------------
// Kernel 2: s1[row] = h[row]·a1, s2[row] = h[row]·a2 (f32). One wave per row.
// ---------------------------------------------------------------------------
__global__ __launch_bounds__(256) void k_scores(const float* __restrict__ h,
                                                const float* __restrict__ a,
                                                float* __restrict__ s1,
                                                float* __restrict__ s2) {
    const int lane = threadIdx.x & 31;
    const int wv   = threadIdx.x >> 5;
    const int row  = blockIdx.x * 8 + wv;   // 0..16383

    const float* __restrict__ hr = h + (size_t)row * FF;
    float p1 = 0.f, p2 = 0.f;
    #pragma unroll
    for (int f = lane; f < FF; f += 32) {
        const float hv = hr[f];
        p1 += hv * a[f];
        p2 += hv * a[FF + f];
    }
    #pragma unroll
    for (int off = 16; off > 0; off >>= 1) {
        p1 += __shfl_xor(p1, off, 32);
        p2 += __shfl_xor(p2, off, 32);
    }
    if (lane == 0) { s1[row] = p1; s2[row] = p2; }
}

// ---------------------------------------------------------------------------
// Kernel 3: per-batch stats. leaky_relu is monotone, so
//   min(e) = lrelu(min s1 + min s2), max(e) = lrelu(max s1 + max s2).
// stats[b] = { mn, 30/(mx-mn) }.
// ---------------------------------------------------------------------------
__global__ __launch_bounds__(256) void k_stats(const float* __restrict__ s1,
                                               const float* __restrict__ s2,
                                               float* __restrict__ stats) {
    __shared__ float sm[4][256];
    const int b = blockIdx.x;
    const int t = threadIdx.x;
    const float INF = __builtin_huge_valf();
    float mn1 = INF, mx1 = -INF, mn2 = INF, mx2 = -INF;
    const float* __restrict__ p1 = s1 + b * NN;
    const float* __restrict__ p2 = s2 + b * NN;
    for (int i = t; i < NN; i += 256) {
        const float v1 = p1[i], v2 = p2[i];
        mn1 = fminf(mn1, v1); mx1 = fmaxf(mx1, v1);
        mn2 = fminf(mn2, v2); mx2 = fmaxf(mx2, v2);
    }
    sm[0][t] = mn1; sm[1][t] = mx1; sm[2][t] = mn2; sm[3][t] = mx2;
    __syncthreads();
    for (int s = 128; s > 0; s >>= 1) {
        if (t < s) {
            sm[0][t] = fminf(sm[0][t], sm[0][t + s]);
            sm[1][t] = fmaxf(sm[1][t], sm[1][t + s]);
            sm[2][t] = fminf(sm[2][t], sm[2][t + s]);
            sm[3][t] = fmaxf(sm[3][t], sm[3][t + s]);
        }
        __syncthreads();
    }
    if (t == 0) {
        const float mn = lrelu(sm[0][0] + sm[2][0]);
        const float mx = lrelu(sm[1][0] + sm[3][0]);
        stats[b * 2]     = mn;
        stats[b * 2 + 1] = 30.f / (mx - mn);
    }
}

// ---------------------------------------------------------------------------
// Kernel 4: materialize f16 attention (A-operand of the big GEMM), computed
// once per element. One v8h (8 j's) per thread.
// ---------------------------------------------------------------------------
__global__ __launch_bounds__(256) void k_att16(const float* __restrict__ s1,
                                               const float* __restrict__ s2,
                                               const float* __restrict__ stats,
                                               _Float16* __restrict__ att16) {
    const size_t idx = (size_t)blockIdx.x * blockDim.x + threadIdx.x;  // B*N*N/8
    const int    j8  = (int)(idx & (NN / 8 - 1));
    const size_t row = idx >> 8;           // NN/8 = 256
    const int    i   = (int)(row & (NN - 1));
    const int    b   = (int)(row >> 11);

    const float s1v = s1[b * NN + i];
    const float mn  = stats[b * 2];
    const float inv = stats[b * 2 + 1];
    const float* __restrict__ s2p = s2 + b * NN + j8 * 8;

    v8h o;
    #pragma unroll
    for (int c = 0; c < 8; ++c) {
        const float x = lrelu(s1v + s2p[c]);
        o[c] = (_Float16)sigm((x - mn) * inv - 20.f);
    }
    ((v8h*)att16)[idx] = o;
}

// ---------------------------------------------------------------------------
// Kernel 5: h_prime = elu(attention @ h) via f16 WMMA 16x16x32, f32 accum.
// A = att16[b][i][j] (row major), B = h16T[b][f][j] (transposed h).
// Per-lane operand fetch = four global_load_b128 per WMMA (ISA §7.12.2):
//   A 16-bit 16x32: e<8 -> K=e+8*half, e>=8 -> K=e+8+8*half
//   B 16-bit 32x16: same K order as A
// One wave per 16(i) x 16(f) tile; K loop = 2048 in steps of 32.
// ---------------------------------------------------------------------------
__global__ __launch_bounds__(256) void k_attn_matmul(const _Float16* __restrict__ att16,
                                                     const _Float16* __restrict__ h16T,
                                                     float* __restrict__ hp) {
    const int lane = threadIdx.x & 31;
    const int wv   = threadIdx.x >> 5;
    const int ti   = blockIdx.x * 8 + wv;   // 0..127 row tiles
    const int tf   = blockIdx.y;            // 0..15 col tiles
    const int b    = blockIdx.z;            // 0..7
    const int half = lane >> 4;
    const int l15  = lane & 15;

    const int i = ti * 16 + l15;            // A-matrix M row for this lane
    const int f = tf * 16 + l15;            // B/D N col for this lane

    const _Float16* __restrict__ aRow = att16 + ((size_t)b * NN + i) * NN;
    const _Float16* __restrict__ bRow = h16T + ((size_t)(b * FF + f)) * NN;

    v8f c = {};
    for (int j0 = 0; j0 < NN; j0 += 32) {
        const v8h alo = *(const v8h*)(aRow + j0 + 8 * half);
        const v8h ahi = *(const v8h*)(aRow + j0 + 16 + 8 * half);
        const v8h blo = *(const v8h*)(bRow + j0 + 8 * half);
        const v8h bhi = *(const v8h*)(bRow + j0 + 16 + 8 * half);
        v16h a, bb;
        #pragma unroll
        for (int e = 0; e < 8; ++e) {
            a[e]  = alo[e];  a[e + 8]  = ahi[e];
            bb[e] = blo[e];  bb[e + 8] = bhi[e];
        }
        c = __builtin_amdgcn_wmma_f32_16x16x32_f16(false, a, false, bb,
                                                   (short)0, c, false, false);
    }

    const int mbase = ti * 16 + half * 8;
    #pragma unroll
    for (int r = 0; r < 8; ++r) {
        float v = c[r];
        v = v > 0.f ? v : (__expf(v) - 1.f);   // ELU, alpha=1
        hp[((size_t)b * NN + mbase + r) * FF + f] = v;
    }
}

// ---------------------------------------------------------------------------
// Kernel 6: final f32 attention output (exact, from f32 s1/s2), overwrites the
// region that temporarily held att16. One float4 (4 j's) per thread.
// ---------------------------------------------------------------------------
__global__ __launch_bounds__(256) void k_write_att(const float* __restrict__ s1,
                                                   const float* __restrict__ s2,
                                                   const float* __restrict__ stats,
                                                   float* __restrict__ att) {
    const size_t idx = (size_t)blockIdx.x * blockDim.x + threadIdx.x;  // B*N*N/4
    const int    j4  = (int)(idx & (NN / 4 - 1));
    const size_t row = idx >> 9;           // NN/4 = 512
    const int    i   = (int)(row & (NN - 1));
    const int    b   = (int)(row >> 11);

    const float s1v = s1[b * NN + i];
    const float mn  = stats[b * 2];
    const float inv = stats[b * 2 + 1];
    const v4f s2v = ((const v4f*)(s2 + b * NN))[j4];

    v4f o;
    #pragma unroll
    for (int c = 0; c < 4; ++c) {
        const float x = lrelu(s1v + s2v[c]);
        o[c] = sigm((x - mn) * inv - 20.f);
    }
    ((v4f*)att)[idx] = o;
}

// ---------------------------------------------------------------------------
extern "C" void kernel_launch(void* const* d_in, const int* in_sizes, int n_in,
                              void* d_out, int out_size, void* d_ws, size_t ws_size,
                              hipStream_t stream) {
    const float* inp = (const float*)d_in[0];   // (8,2048,256)
    // d_in[1] = adj: unused by the reference computation
    const float* W   = (const float*)d_in[2];   // (256,256)
    const float* a   = (const float*)d_in[3];   // (512,1)

    float* out = (float*)d_out;
    float* hp  = out;                                    // (8,2048,256)
    float* att = out + (size_t)BATCH * NN * FF;          // (8,2048,2048) f32
    // att16 temporarily lives at the head of the f32 attention region; it is
    // consumed by k_attn_matmul and then overwritten by k_write_att (stream
    // order guarantees safety, every byte rewritten each call).
    _Float16* att16 = (_Float16*)att;

    char*     ws    = (char*)d_ws;
    float*    h32   = (float*)ws;                                        // 16 MB
    _Float16* h16T  = (_Float16*)(ws + (size_t)BATCH * NN * FF * 4);     //  8 MB
    float*    s1    = (float*)(ws + (size_t)BATCH * NN * FF * 6);
    float*    s2    = s1 + BATCH * NN;
    float*    stats = s2 + BATCH * NN;                                   // 8 x {mn, inv}

    k_gemm_h     <<<dim3(128, 16), 256, 0, stream>>>(inp, W, h32, h16T);
    k_scores     <<<2048,          256, 0, stream>>>(h32, a, s1, s2);
    k_stats      <<<BATCH,         256, 0, stream>>>(s1, s2, stats);
    k_att16      <<<16384,         256, 0, stream>>>(s1, s2, stats, att16);
    k_attn_matmul<<<dim3(16, 16, BATCH), 256, 0, stream>>>(att16, h16T, hp);
    k_write_att  <<<32768,         256, 0, stream>>>(s1, s2, stats, att);
}
